// Dynamic_conv3d_23072564314510
// MI455X (gfx1250) — matrix operand order repaired and checked by reading, not compiled, not run on hardware
//
#include <hip/hip_runtime.h>
#include <hip/hip_bf16.h>

typedef __attribute__((ext_vector_type(16))) __bf16 v16bf;
typedef __attribute__((ext_vector_type(8)))  float  v8f;

__device__ __forceinline__ unsigned short f2bf(float f) {
    unsigned int u = __builtin_bit_cast(unsigned int, f);
    unsigned int r = (u + 0x7FFFu + ((u >> 16) & 1u)) >> 16;
    return (unsigned short)r;
}

// A-fragment K permutation within a 32-K group (involution: swaps 8-15 <-> 16-23).
__device__ __forceinline__ int kperm(int r) {
    if (r < 8)  return r;
    if (r < 16) return r + 8;
    if (r < 24) return r - 8;
    return r;
}

// ---------------------------------------------------------------------------
// Kernel A: per-(b,c) pooling.  p[b,c,d] = mean_{h,w} x ; s[b,c] = mean_{d,h,w} x
// 256 blocks (b*32+c), 256 threads. Deterministic tree reduction, no atomics.
// ---------------------------------------------------------------------------
__global__ __launch_bounds__(256)
void pool_kernel(const float* __restrict__ x, float* __restrict__ p, float* __restrict__ s) {
    const int bc = blockIdx.x;
    const float* base = x + (size_t)bc * 16 * 9216;
    __shared__ float red[256];
    float tot = 0.f;
    for (int d = 0; d < 16; ++d) {
        float acc = 0.f;
        const float* pd = base + d * 9216;
        for (int i = threadIdx.x; i < 9216; i += 256) acc += pd[i];
        red[threadIdx.x] = acc;
        __syncthreads();
        for (int off = 128; off > 0; off >>= 1) {
            if (threadIdx.x < off) red[threadIdx.x] += red[threadIdx.x + off];
            __syncthreads();
        }
        if (threadIdx.x == 0) { p[bc * 16 + d] = red[0] * (1.f / 9216.f); tot += red[0]; }
        __syncthreads();
    }
    if (threadIdx.x == 0) s[bc] = tot * (1.f / (16.f * 9216.f));
}

// ---------------------------------------------------------------------------
// Kernel B: attention MLP + softmax, 1D conv + BN + sigmoid gate, and agg_w
// in tap-major layout  aggw[b][t=27][m=32][c=32] (bf16), with c kept in
// natural order (no K permutation) so the conv kernel can load A
// fragments as contiguous 32B per-lane global reads.
// ---------------------------------------------------------------------------
__global__ __launch_bounds__(256)
void prep_kernel(const float* __restrict__ s, const float* __restrict__ p,
                 const float* __restrict__ fc1, const float* __restrict__ fc2,
                 const float* __restrict__ naw, const float* __restrict__ nab,
                 const float* __restrict__ gamma, const float* __restrict__ beta,
                 const float* __restrict__ mean, const float* __restrict__ var,
                 const float* __restrict__ nbw, const float* __restrict__ weight,
                 float* __restrict__ a_ws, float* __restrict__ gate,
                 unsigned short* __restrict__ aggw) {
    __shared__ float hbuf[8 * 129];
    __shared__ float att[8 * 4];
    const int t = threadIdx.x;

    // h = relu(s @ fc1^T)  [8,129]
    for (int i = t; i < 8 * 129; i += 256) {
        int b = i / 129, j = i % 129;
        float acc = 0.f;
        for (int c = 0; c < 32; ++c) acc += s[b * 32 + c] * fc1[j * 32 + c];
        hbuf[i] = fmaxf(acc, 0.f);
    }
    __syncthreads();

    // logits/T -> softmax  [8,4]
    if (t < 8) {
        float lg[4]; float mx = -1e30f;
        for (int k = 0; k < 4; ++k) {
            float acc = 0.f;
            for (int j = 0; j < 129; ++j) acc += hbuf[t * 129 + j] * fc2[k * 129 + j];
            lg[k] = acc * 0.25f;                 // /TEMP
            mx = fmaxf(mx, lg[k]);
        }
        float sum = 0.f;
        for (int k = 0; k < 4; ++k) { lg[k] = __expf(lg[k] - mx); sum += lg[k]; }
        float inv = 1.f / sum;
        for (int k = 0; k < 4; ++k) att[t * 4 + k] = lg[k] * inv;
    }
    __syncthreads();

    // a = relu(BN(conv1d(p, naw) + nab)) -> a_ws [8][128][16]
    for (int i = t; i < 8 * 128 * 16; i += 256) {
        int d = i & 15, o = (i >> 4) & 127, b = i >> 11;
        float acc = nab[o];
        for (int c = 0; c < 32; ++c) {
            const float* pr = p + (b * 32 + c) * 16;
            const float* wr = naw + (o * 32 + c) * 3;
            #pragma unroll
            for (int u = 0; u < 3; ++u) {
                int dd = d + u - 1;
                if ((unsigned)dd < 16u) acc += pr[dd] * wr[u];
            }
        }
        float inv = rsqrtf(var[o] + 1e-3f);
        float v = (acc - mean[o]) * inv * gamma[o] + beta[o];
        a_ws[i] = fmaxf(v, 0.f);
    }
    __syncthreads();

    // gate = sigmoid(nbw @ a)  [8][32][16]
    for (int i = t; i < 8 * 32 * 16; i += 256) {
        int d = i & 15, co = (i >> 4) & 31, b = i >> 9;
        float acc = 0.f;
        const float* ab = a_ws + b * 128 * 16;
        const float* wr = nbw + co * 128;
        for (int o = 0; o < 128; ++o) acc += wr[o] * ab[o * 16 + d];
        gate[i] = 1.f / (1.f + __expf(-acc));
    }

    // aggw[b][tap][m][slot] , slot = c;  weight is [4][32][32][27]
    for (int i = t; i < 8 * 27 * 32 * 32; i += 256) {
        int slot = i & 31;
        int m    = (i >> 5) & 31;
        int tap  = (i >> 10) % 27;
        int b    = i / (27 * 1024);
        int c    = slot;
        float acc = 0.f;
        #pragma unroll
        for (int k = 0; k < 4; ++k)
            acc += att[b * 4 + k] * weight[((size_t)(k * 32 + m) * 32 + c) * 27 + tap];
        aggw[i] = f2bf(acc);
    }
}

// ---------------------------------------------------------------------------
// Kernel C: fused gate*x + 3x3x3 conv as 27 tap-GEMMs (K=32 channels) in bf16
// WMMA.  Block = 256 threads (8 waves) -> output tile [Cout=32][h0..h0+1][32 w]
// for one (b,d).  LDS holds the raw gated halo tile Y[3 dz][4 hy][34 w][32 c]
// (bf16, 25.5 KB); B fragments for each tap are shifted views of it (c = K,
// contiguous, 32B-aligned).  A fragments stream from L2.
// ---------------------------------------------------------------------------
__global__ __launch_bounds__(256)
void conv_wmma_kernel(const float* __restrict__ x, const float* __restrict__ gate,
                      const unsigned short* __restrict__ aggw, float* __restrict__ out) {
    const int tid = threadIdx.x;
    const int b   = blockIdx.z;
    const int d   = blockIdx.y;
    const int h0  = (blockIdx.x / 3) * 2;        // 48 h-tiles of 2 rows
    const int w0  = (blockIdx.x % 3) * 32;       // 3 w-tiles of 32

    __shared__ __align__(32) unsigned short Ys[3 * 4 * 34 * 32];   // 26,112 B

    const float* xb    = x + (size_t)b * 32 * 16 * 9216;
    const float* gateb = gate + b * 32 * 16;
    const unsigned short* aggwb = aggw + (size_t)b * 27 * 32 * 32;

    // Warm L2/L1 for this sample's tap-major A matrix (global_prefetch_b8).
    if (tid < 32) __builtin_prefetch(aggwb + (size_t)tid * 864, 0, 3);

    // ---- fill gated halo tile: 32c x 3dz x 4hy x 34w, each x element once ----
    for (int i = tid; i < 32 * 3 * 4 * 34; i += 256) {      // 13,056 items, 51/thread
        int wloc  = i % 34;
        int combo = i / 34;                                  // (c, dz, hyloc)
        int hyloc = combo & 3;
        int dz    = (combo >> 2) % 3;
        int c     = combo / 12;
        int zz = d  + dz    - 1;
        int yy = h0 + hyloc - 1;
        int xx = w0 + wloc  - 1;
        bool zok = (unsigned)zz < 16u;
        float v = 0.f;
        if (zok && (unsigned)yy < 96u && (unsigned)xx < 96u)
            v = xb[(((size_t)c * 16 + zz) * 96 + yy) * 96 + xx] * gateb[c * 16 + zz];
        Ys[((dz * 4 + hyloc) * 34 + wloc) * 32 + c] = f2bf(v);
    }
    __syncthreads();

    // ---- 27 tap-GEMMs: [16m x 32c] x [32c x 16n] accumulated in f32 ----
    const int wave  = tid >> 5;
    const int lane  = tid & 31;
    const int hsub  = wave >> 2;          // 0..1 : output h row
    const int mtile = (wave >> 1) & 1;    // 0..1 : cout 0-15 / 16-31
    const int ntile = wave & 1;           // 0..1 : w sub-tile
    const int khalf = lane >> 4;          // c half (K) within the fragment
    const int l16   = lane & 15;
    const int mrow  = mtile * 16 + l16;

    const unsigned short* arow = aggwb + (size_t)mrow * 32 + khalf * 16;
    const int bcol = ntile * 16 + l16;    // output n -> w offset base

    v8f acc = {0.f, 0.f, 0.f, 0.f, 0.f, 0.f, 0.f, 0.f};
    #pragma unroll
    for (int t = 0; t < 27; ++t) {
        const int dz = t / 9, dy = (t % 9) / 3, dx = t % 3;
        v16bf af = *(const v16bf*)(arow + (size_t)t * 1024);
        v16bf bf = *(const v16bf*)(&Ys[((dz * 4 + (hsub + dy)) * 34 + bcol + dx) * 32
                                       + khalf * 16]);
        acc = __builtin_amdgcn_wmma_f32_16x16x32_bf16(
            /*neg_a=*/false, af, /*neg_b=*/false, bf,
            /*c_mod=*/(short)0, acc, /*reuse_a=*/false, /*reuse_b=*/false);
    }

    // ---- store: C/D layout — VGPR r: lanes 0-15 M=r, lanes 16-31 M=8+r; N=lane%16
    const int hh = h0 + hsub;
    const int w  = w0 + ntile * 16 + l16;
    const int mb = mtile * 16 + khalf * 8;
    #pragma unroll
    for (int r = 0; r < 8; ++r) {
        int cout = mb + r;
        out[(((size_t)(b * 32 + cout) * 16 + d) * 96 + hh) * 96 + w] = acc[r];
    }
}

extern "C" void kernel_launch(void* const* d_in, const int* in_sizes, int n_in,
                              void* d_out, int out_size, void* d_ws, size_t ws_size,
                              hipStream_t stream) {
    (void)in_sizes; (void)n_in; (void)out_size; (void)ws_size;
    const float* x    = (const float*)d_in[0];   // [8,32,16,96,96]
    const float* fc1  = (const float*)d_in[1];   // [129,32]
    const float* fc2  = (const float*)d_in[2];   // [4,129]
    const float* naw  = (const float*)d_in[3];   // [128,32,3]
    const float* nab  = (const float*)d_in[4];   // [128]
    const float* gam  = (const float*)d_in[5];   // [128]
    const float* bet  = (const float*)d_in[6];   // [128]
    const float* mea  = (const float*)d_in[7];   // [128]
    const float* var  = (const float*)d_in[8];   // [128]
    const float* nbw  = (const float*)d_in[9];   // [32,128]
    const float* wgt  = (const float*)d_in[10];  // [4,32,32,3,3,3]

    // workspace layout (all 32B aligned)
    float* p_ws  = (float*)d_ws;               // 4096 f : p[8][32][16]
    float* s_ws  = p_ws + 4096;                //  256 f : s[8][32]
    float* a_ws  = s_ws + 256;                 // 16384 f: a[8][128][16]
    float* g_ws  = a_ws + 16384;               // 4096 f : gate[8][32][16]
    unsigned short* aggw = (unsigned short*)(g_ws + 4096);   // 8*27*32*32 bf16

    pool_kernel<<<256, 256, 0, stream>>>(x, p_ws, s_ws);
    prep_kernel<<<1, 256, 0, stream>>>(s_ws, p_ws, fc1, fc2, naw, nab,
                                       gam, bet, mea, var, nbw, wgt,
                                       a_ws, g_ws, aggw);
    dim3 grid(48 * 3, 16, 8);
    conv_wmma_kernel<<<grid, 256, 0, stream>>>(x, g_ws, aggw, (float*)d_out);
}
